// EncoderDecoder_60705067761839
// MI455X (gfx1250) — matrix operand order repaired and checked by reading, not compiled, not run on hardware
//
#include <hip/hip_runtime.h>

#define USE_ASYNC_LDS 1

typedef __bf16 bf16_t;
typedef __attribute__((ext_vector_type(16))) __bf16 v16bf;
typedef __attribute__((ext_vector_type(8)))  __bf16 v8bf;
typedef __attribute__((ext_vector_type(8)))  float  v8f;

__device__ __forceinline__ float sigmf(float v) { return 1.f / (1.f + __expf(-v)); }

#define WMMA_BF16(acc, a, b)                                                   \
    acc = __builtin_amdgcn_wmma_f32_16x16x32_bf16(false, (a), false, (b),      \
                                                  (short)0, (acc), false, false)

// ---------------------------------------------------------------------------
// Async global->LDS copy (ASYNCcnt path). 16B granularity, nbytes % 16 == 0.
// ---------------------------------------------------------------------------
__device__ __forceinline__ void async_copy_g2l(bf16_t* dst_lds, const bf16_t* src_g,
                                               int nbytes, int tid, int nthr)
{
#if USE_ASYNC_LDS
    for (int off = tid * 16; off < nbytes; off += nthr * 16) {
        unsigned lds = (unsigned)(uintptr_t)dst_lds + (unsigned)off;
        unsigned long long ga = (unsigned long long)(uintptr_t)src_g + (unsigned)off;
        asm volatile("global_load_async_to_lds_b128 %0, %1, off"
                     :: "v"(lds), "v"(ga) : "memory");
    }
    asm volatile("s_wait_asynccnt 0x0" ::: "memory");
#else
    for (int off = tid * 16; off < nbytes; off += nthr * 16)
        *(v8bf*)((char*)dst_lds + off) = *(const v8bf*)((const char*)src_g + off);
#endif
}

// A-fragment from a single source: 16x32 bf16, contiguous k layout (same as B).
// Row m; lanes 0-15 K = k0+{0..15}; lanes 16-31 K = k0+{16..31} (kbase = k0 + lhalf*8).
__device__ __forceinline__ v16bf afrag2(const bf16_t* base, int stride, int m, int kbase)
{
    const v8bf lo = *(const v8bf*)(base + (size_t)m * stride + kbase + (kbase & 8));
    const v8bf hi = *(const v8bf*)(base + (size_t)m * stride + kbase + (kbase & 8) + 8);
    return __builtin_shufflevector(lo, hi, 0, 1, 2, 3, 4, 5, 6, 7,
                                   8, 9, 10, 11, 12, 13, 14, 15);
}

// ---------------------------------------------------------------------------
// Legacy gather-based GEMM (one-shot bridge GEMM only).
// ---------------------------------------------------------------------------
template <typename AF>
__device__ __forceinline__ void gemm_wmma_gather(AF afetch, const bf16_t* __restrict__ W,
                                                 int Kt, int NG,
                                                 float* __restrict__ gout, int gld)
{
    const int tid = threadIdx.x, lane = tid & 31, wave = tid >> 5;
    const int nwaves = blockDim.x >> 5;
    const int lm = lane & 15, lhalf = (lane >> 4) & 1;
    const int ntiles = 2 * (NG >> 4);
    for (int tile = wave; tile < ntiles; tile += nwaves) {
        const int mt = tile & 1, nt = tile >> 1;
        const int m = mt * 16 + lm, n = nt * 16 + lm;
        v8f acc = {};
        for (int k0 = 0; k0 < Kt; k0 += 32) {
            v16bf a;
            const int kbase = k0 + lhalf * 8;
            #pragma unroll
            for (int i = 0; i < 16; ++i)
                a[i] = (bf16_t)afetch(m, kbase + lhalf * 8 + i);
            const v16bf b = *(const v16bf*)(W + (size_t)n * Kt + k0 + lhalf * 16);
            WMMA_BF16(acc, a, b);
        }
        #pragma unroll
        for (int v = 0; v < 8; ++v)
            gout[(size_t)(mt * 16 + v + lhalf * 8) * gld + nt * 16 + lm] = acc[v];
    }
}

// ---------------------------------------------------------------------------
// Fused GRU step: gates GEMM (A=[x|h] 32xKt, W=Wcat 2048xKt) with the gate
// nonlinearity fused in the epilogue. Wave w owns j-columns [w*32, w*32+32).
// K-loop is split into the input region (A from inb) and hidden region
// (A from hbuf, B offset din) -> fully affine addressing, no per-iter selects.
// Requires din % 32 == 0 (64/128/512/1024 all qualify).
// ---------------------------------------------------------------------------
__device__ __forceinline__ void gru_step_fused(const bf16_t* inb, int din,
                                               bf16_t* hbuf, int hstride,
                                               const bf16_t* __restrict__ W,
                                               const float bias_r[2], const float bias_z[2],
                                               const float bias_gi[2], const float bias_gh[2],
                                               bf16_t* seq_out_row /* or nullptr */,
                                               int out_stride, int col_off)
{
    const int tid = threadIdx.x, lane = tid & 31, wave = tid >> 5;
    const int lm = lane & 15, lhalf = (lane >> 4) & 1;
    const int jt0 = wave * 2;
    const int Kt = din + 512;

    v8f acc[4][2][2] = {};

    // --- phase 1: input region (k in [0, din)) ---
    for (int k0 = 0; k0 < din; k0 += 32) {
        const int kbase = k0 + lhalf * 8;
        const v16bf a0 = afrag2(inb, din, lm, kbase);
        const v16bf a1 = afrag2(inb, din, 16 + lm, kbase);
        #pragma unroll
        for (int g = 0; g < 4; ++g) {
            #pragma unroll
            for (int c = 0; c < 2; ++c) {
                const int n = g * 512 + (jt0 + c) * 16 + lm;
                const v16bf b = *(const v16bf*)(W + (size_t)n * Kt + k0 + lhalf * 16);
                WMMA_BF16(acc[g][c][0], a0, b);
                WMMA_BF16(acc[g][c][1], a1, b);
            }
        }
    }
    // --- phase 2: hidden region (k in [din, din+512)) ---
    for (int k0 = 0; k0 < 512; k0 += 32) {
        const int kbase = k0 + lhalf * 8;
        const v16bf a0 = afrag2(hbuf, hstride, lm, kbase);
        const v16bf a1 = afrag2(hbuf, hstride, 16 + lm, kbase);
        #pragma unroll
        for (int g = 0; g < 4; ++g) {
            #pragma unroll
            for (int c = 0; c < 2; ++c) {
                const int n = g * 512 + (jt0 + c) * 16 + lm;
                const v16bf b = *(const v16bf*)(W + (size_t)n * Kt + din + k0 + lhalf * 16);
                WMMA_BF16(acc[g][c][0], a0, b);
                WMMA_BF16(acc[g][c][1], a1, b);
            }
        }
    }

    __syncthreads();  // all hbuf reads complete before updates
    #pragma unroll
    for (int c = 0; c < 2; ++c) {
        const int j = (jt0 + c) * 16 + lm;
        #pragma unroll
        for (int mt = 0; mt < 2; ++mt) {
            #pragma unroll
            for (int v = 0; v < 8; ++v) {
                const int row = mt * 16 + v + lhalf * 8;
                const float r  = sigmf(acc[0][c][mt][v] + bias_r[c]);
                const float z  = sigmf(acc[1][c][mt][v] + bias_z[c]);
                const float nn = tanhf(acc[2][c][mt][v] + bias_gi[c] +
                                       r * (acc[3][c][mt][v] + bias_gh[c]));
                const float hp = (float)hbuf[(size_t)row * hstride + j];
                const float h2 = (1.f - z) * nn + z * hp;
                hbuf[(size_t)row * hstride + j] = (bf16_t)h2;
                if (seq_out_row)
                    seq_out_row[(size_t)row * out_stride + col_off + j] = (bf16_t)h2;
            }
        }
    }
}

// ---------------------------------------------------------------------------
// Weight prep: fused-gate Wcat (2048 x (din+512)) bf16 + bias_cat (2048).
// rows [0,1024): r,z [Wih|Whh]; [1024,1536): [Wih_n|0]; [1536,2048): [0|Whh_n].
// ---------------------------------------------------------------------------
__global__ void prep_wcat_kernel(const float* __restrict__ Wih, const float* __restrict__ Whh,
                                 const float* __restrict__ bih, const float* __restrict__ bhh,
                                 int din, bf16_t* __restrict__ Wcat, float* __restrict__ bias)
{
    const int Kt = din + 512;
    const size_t total = (size_t)2048 * Kt;
    const size_t stride = (size_t)gridDim.x * blockDim.x;
    for (size_t i = (size_t)blockIdx.x * blockDim.x + threadIdx.x; i < total; i += stride) {
        const int rr = (int)(i / Kt);
        const int c  = (int)(i % Kt);
        float v = 0.f;
        if (rr < 1024)      v = (c < din) ? Wih[(size_t)rr * din + c]
                                          : Whh[(size_t)rr * 512 + (c - din)];
        else if (rr < 1536) { if (c < din) v = Wih[(size_t)rr * din + c]; }
        else                { if (c >= din) v = Whh[(size_t)(rr - 512) * 512 + (c - din)]; }
        Wcat[i] = (bf16_t)v;
    }
    for (int rr = blockIdx.x * blockDim.x + threadIdx.x; rr < 2048;
         rr += gridDim.x * blockDim.x) {
        float bv;
        if (rr < 1024)      bv = bih[rr] + bhh[rr];
        else if (rr < 1536) bv = bih[rr];
        else                bv = bhh[rr - 512];
        bias[rr] = bv;
    }
}

__global__ void f32_to_bf16_kernel(const float* __restrict__ src,
                                   bf16_t* __restrict__ dst, size_t n)
{
    const size_t stride = (size_t)gridDim.x * blockDim.x;
    for (size_t i = (size_t)blockIdx.x * blockDim.x + threadIdx.x; i < n; i += stride)
        dst[i] = (bf16_t)src[i];
}

// x (32,256,128) f32 -> seq0 (256,32,128) bf16
__global__ void prep_x_kernel(const float* __restrict__ x, bf16_t* __restrict__ seq0)
{
    const size_t total = (size_t)32 * 256 * 128;
    const size_t stride = (size_t)gridDim.x * blockDim.x;
    for (size_t i = (size_t)blockIdx.x * blockDim.x + threadIdx.x; i < total; i += stride) {
        const int k = (int)(i % 128);
        const size_t r = i / 128;
        const int t = (int)(r % 256), b = (int)(r / 256);
        seq0[((size_t)t * 32 + b) * 128 + k] = (bf16_t)x[i];
    }
}

// y (32,64,64) f32 -> ybf (64,32,64) bf16
__global__ void prep_y_kernel(const float* __restrict__ y, bf16_t* __restrict__ ybf)
{
    const size_t total = (size_t)32 * 64 * 64;
    const size_t stride = (size_t)gridDim.x * blockDim.x;
    for (size_t i = (size_t)blockIdx.x * blockDim.x + threadIdx.x; i < total; i += stride) {
        const int k = (int)(i % 64);
        const size_t r = i / 64;
        const int u = (int)(r % 64), b = (int)(r / 64);
        ybf[((size_t)u * 32 + b) * 64 + k] = (bf16_t)y[i];
    }
}

// ---------------------------------------------------------------------------
// One bidirectional GRU layer. grid.x = 2 (0=fwd,1=bwd), block = 512 (16 waves).
// seq_in (256,32,din) bf16; seq_out (256,32,1024) bf16 (fwd cols 0..511, bwd 512..1023).
// ---------------------------------------------------------------------------
__global__ __launch_bounds__(512)
void gru_layer_kernel(const bf16_t* __restrict__ seq_in, int din,
                      bf16_t* __restrict__ seq_out,
                      const bf16_t* __restrict__ Wcat_f, const float* __restrict__ bias_f,
                      const bf16_t* __restrict__ Wcat_b, const float* __restrict__ bias_b,
                      float* __restrict__ hfin_f, float* __restrict__ hfin_b)
{
    __shared__ bf16_t xbuf[32 * 1024];
    __shared__ bf16_t hbuf[32 * 512];
    const int dir = blockIdx.x;
    const bf16_t* W   = dir ? Wcat_b : Wcat_f;
    const float* bias = dir ? bias_b : bias_f;
    float* hfin       = dir ? hfin_b : hfin_f;
    const int col_off = dir ? 512 : 0;
    const int tid = threadIdx.x, lane = tid & 31, wave = tid >> 5;
    const int lm = lane & 15;
    const int jt0 = wave * 2;

    float bias_r[2], bias_z[2], bias_gi[2], bias_gh[2];
    #pragma unroll
    for (int c = 0; c < 2; ++c) {
        const int j = (jt0 + c) * 16 + lm;
        bias_r[c]  = bias[j];
        bias_z[c]  = bias[512 + j];
        bias_gi[c] = bias[1024 + j];
        bias_gh[c] = bias[1536 + j];
    }
    for (int i = tid; i < 32 * 512; i += 512) hbuf[i] = (bf16_t)0.f;

    for (int s = 0; s < 256; ++s) {
        const int t = dir ? (255 - s) : s;
        async_copy_g2l(xbuf, seq_in + (size_t)t * 32 * din, 32 * din * 2, tid, 512);
        __syncthreads();
        gru_step_fused(xbuf, din, hbuf, 512, W, bias_r, bias_z, bias_gi, bias_gh,
                       seq_out + (size_t)t * 32 * 1024, 1024, col_off);
    }
    __syncthreads();
    for (int i = tid; i < 32 * 512; i += 512) hfin[i] = (float)hbuf[i];
}

// dec_state (32x1536) = [hb | hf] @ enc_lin^T + enc_lin_b (one-shot bridge)
__global__ __launch_bounds__(512)
void dec_init_kernel(const float* __restrict__ hb, const float* __restrict__ hf,
                     const bf16_t* __restrict__ Wlin, const float* __restrict__ blin,
                     float* __restrict__ dec_state)
{
    auto afetch = [&](int m, int k) -> float {
        return (k < 512) ? hb[m * 512 + k] : hf[m * 512 + (k - 512)];
    };
    gemm_wmma_gather(afetch, Wlin, 1024, 1536, dec_state, 1536);
    __threadfence();
    __syncthreads();
    for (int idx = threadIdx.x; idx < 32 * 1536; idx += blockDim.x)
        dec_state[idx] += blin[idx % 1536];
}

// ---------------------------------------------------------------------------
// Decoder: 3 chained GRU layers + logits + log-softmax loss. Single workgroup.
// ---------------------------------------------------------------------------
__global__ __launch_bounds__(512)
void decoder_kernel(const float* __restrict__ y,       // (32,64,64) f32 (for loss)
                    const bf16_t* __restrict__ ybf,    // (64,32,64) bf16 (teacher input)
                    const float* __restrict__ dstate,  // (32,1536) f32 initial states
                    const bf16_t* __restrict__ Wc0, const float* __restrict__ bb0,
                    const bf16_t* __restrict__ Wc1, const float* __restrict__ bb1,
                    const bf16_t* __restrict__ Wc2, const float* __restrict__ bb2,
                    const bf16_t* __restrict__ WL,  const float* __restrict__ bL,
                    float* __restrict__ out)
{
    __shared__ bf16_t st[3][32 * 512];
    __shared__ bf16_t ybuf[32 * 64];
    __shared__ float  logits_sh[32 * 64];
    const int tid = threadIdx.x, lane = tid & 31, wave = tid >> 5;
    const int lm = lane & 15, lhalf = (lane >> 4) & 1;
    const int jt0 = wave * 2;

    float br[3][2], bz[3][2], bgi[3][2], bgh[3][2];
    const float* bbs[3] = {bb0, bb1, bb2};
    #pragma unroll
    for (int l = 0; l < 3; ++l)
        #pragma unroll
        for (int c = 0; c < 2; ++c) {
            const int j = (jt0 + c) * 16 + lm;
            br[l][c]  = bbs[l][j];
            bz[l][c]  = bbs[l][512 + j];
            bgi[l][c] = bbs[l][1024 + j];
            bgh[l][c] = bbs[l][1536 + j];
        }
    for (int i = tid; i < 32 * 512; i += 512) {
        const int b = i >> 9, j = i & 511;
        st[0][i] = (bf16_t)dstate[(size_t)b * 1536 + j];
        st[1][i] = (bf16_t)dstate[(size_t)b * 1536 + 512 + j];
        st[2][i] = (bf16_t)dstate[(size_t)b * 1536 + 1024 + j];
    }

    float logp = 0.f;
    const bf16_t* Ws[3] = {Wc0, Wc1, Wc2};

    for (int u = 0; u < 64; ++u) {
        if (u == 0) {
            for (int i = tid; i < 32 * 64; i += 512) ybuf[i] = (bf16_t)0.f;
        } else {
            async_copy_g2l(ybuf, ybf + (size_t)(u - 1) * 32 * 64, 32 * 64 * 2, tid, 512);
        }
        __syncthreads();

        #pragma unroll
        for (int l = 0; l < 3; ++l) {
            const bf16_t* inb = (l == 0) ? ybuf : st[l - 1];
            const int din = (l == 0) ? 64 : 512;
            gru_step_fused(inb, din, st[l], 512, Ws[l],
                           br[l], bz[l], bgi[l], bgh[l], nullptr, 0, 0);
            __syncthreads();
        }

        // logits = st[2] @ WL^T  (32 x 512 x 64): 8 tiles, waves 0-7
        if (wave < 8) {
            const int mt = wave & 1, nt = wave >> 1;
            const int m = mt * 16 + lm, n = nt * 16 + lm;
            v8f acc = {};
            for (int k0 = 0; k0 < 512; k0 += 32) {
                const int kbase = k0 + lhalf * 8;
                const v16bf a = afrag2(st[2], 512, m, kbase);
                const v16bf b = *(const v16bf*)(WL + (size_t)n * 512 + k0 + lhalf * 16);
                WMMA_BF16(acc, a, b);
            }
            #pragma unroll
            for (int v = 0; v < 8; ++v)
                logits_sh[(size_t)(mt * 16 + v + lhalf * 8) * 64 + nt * 16 + lm] = acc[v];
        }
        __syncthreads();

        if (tid < 32) {
            const int b = tid;
            float mx = -1e30f;
            for (int j = 0; j < 64; ++j)
                mx = fmaxf(mx, logits_sh[b * 64 + j] + bL[j]);
            float se = 0.f;
            for (int j = 0; j < 64; ++j)
                se += __expf(logits_sh[b * 64 + j] + bL[j] - mx);
            const float lse = mx + __logf(se);
            float accp = 0.f;
            for (int j = 0; j < 64; ++j)
                accp += (logits_sh[b * 64 + j] + bL[j] - lse) *
                        y[((size_t)b * 64 + u) * 64 + j];
            logp += accp;
        }
        __syncthreads();
    }
    if (tid < 32) out[tid] = logp;
}

// ---------------------------------------------------------------------------
extern "C" void kernel_launch(void* const* d_in, const int* in_sizes, int n_in,
                              void* d_out, int out_size, void* d_ws, size_t ws_size,
                              hipStream_t stream)
{
    (void)in_sizes; (void)n_in; (void)out_size; (void)ws_size;
    const float* x = (const float*)d_in[0];
    const float* y = (const float*)d_in[1];
    const float* encP[3][8];
    for (int l = 0; l < 3; ++l)
        for (int i = 0; i < 8; ++i) encP[l][i] = (const float*)d_in[2 + l * 8 + i];
    const float* enc_lin_W = (const float*)d_in[26];
    const float* enc_lin_b = (const float*)d_in[27];
    const float* decP[3][4];
    for (int l = 0; l < 3; ++l)
        for (int i = 0; i < 4; ++i) decP[l][i] = (const float*)d_in[28 + l * 4 + i];
    const float* dec_lin_W = (const float*)d_in[40];
    const float* dec_lin_b = (const float*)d_in[41];
    float* out = (float*)d_out;

    char* wp = (char*)d_ws;
    auto alloc = [&](size_t bytes) -> void* {
        void* r = (void*)wp;
        wp += (bytes + 255) & ~(size_t)255;
        return r;
    };

    bf16_t* seq0 = (bf16_t*)alloc((size_t)256 * 32 * 128 * 2);
    bf16_t* seq1 = (bf16_t*)alloc((size_t)256 * 32 * 1024 * 2);
    bf16_t* seq2 = (bf16_t*)alloc((size_t)256 * 32 * 1024 * 2);
    bf16_t* ybf  = (bf16_t*)alloc((size_t)64 * 32 * 64 * 2);

    const int dinE[3] = {128, 1024, 1024};
    bf16_t* WcE[3][2]; float* bE[3][2];
    for (int l = 0; l < 3; ++l)
        for (int d = 0; d < 2; ++d) {
            WcE[l][d] = (bf16_t*)alloc((size_t)2048 * (dinE[l] + 512) * 2);
            bE[l][d]  = (float*)alloc(2048 * 4);
        }
    const int dinD[3] = {64, 512, 512};
    bf16_t* WcD[3]; float* bD[3];
    for (int l = 0; l < 3; ++l) {
        WcD[l] = (bf16_t*)alloc((size_t)2048 * (dinD[l] + 512) * 2);
        bD[l]  = (float*)alloc(2048 * 4);
    }
    bf16_t* encLin16 = (bf16_t*)alloc((size_t)1536 * 1024 * 2);
    bf16_t* decLin16 = (bf16_t*)alloc((size_t)64 * 512 * 2);
    float* hfinF = (float*)alloc((size_t)32 * 512 * 4);
    float* hfinB = (float*)alloc((size_t)32 * 512 * 4);
    float* dstate = (float*)alloc((size_t)32 * 1536 * 4);

    // ---- prep ----
    prep_x_kernel<<<256, 256, 0, stream>>>(x, seq0);
    prep_y_kernel<<<64, 256, 0, stream>>>(y, ybf);
    for (int l = 0; l < 3; ++l)
        for (int d = 0; d < 2; ++d)
            prep_wcat_kernel<<<512, 256, 0, stream>>>(
                encP[l][d * 4 + 0], encP[l][d * 4 + 1],
                encP[l][d * 4 + 2], encP[l][d * 4 + 3],
                dinE[l], WcE[l][d], bE[l][d]);
    for (int l = 0; l < 3; ++l)
        prep_wcat_kernel<<<512, 256, 0, stream>>>(
            decP[l][0], decP[l][1], decP[l][2], decP[l][3],
            dinD[l], WcD[l], bD[l]);
    f32_to_bf16_kernel<<<256, 256, 0, stream>>>(enc_lin_W, encLin16, (size_t)1536 * 1024);
    f32_to_bf16_kernel<<<64, 256, 0, stream>>>(dec_lin_W, decLin16, (size_t)64 * 512);

    // ---- encoder: 3 bidirectional layers ----
    gru_layer_kernel<<<2, 512, 0, stream>>>(seq0, 128, seq1,
        WcE[0][0], bE[0][0], WcE[0][1], bE[0][1], hfinF, hfinB);
    gru_layer_kernel<<<2, 512, 0, stream>>>(seq1, 1024, seq2,
        WcE[1][0], bE[1][0], WcE[1][1], bE[1][1], hfinF, hfinB);
    gru_layer_kernel<<<2, 512, 0, stream>>>(seq2, 1024, seq1,
        WcE[2][0], bE[2][0], WcE[2][1], bE[2][1], hfinF, hfinB);

    // ---- bridge + decoder ----
    dec_init_kernel<<<1, 512, 0, stream>>>(hfinB, hfinF, encLin16, enc_lin_b, dstate);
    decoder_kernel<<<1, 512, 0, stream>>>(y, ybf, dstate,
        WcD[0], bD[0], WcD[1], bD[1], WcD[2], bD[2],
        decLin16, dec_lin_b, out);
}
